// STU_10402410791725
// MI455X (gfx1250) — matrix operand order repaired and checked by reading, not compiled, not run on hardware
//
#include <hip/hip_runtime.h>

// ---------------------------------------------------------------------------
// STU spectral layer for MI455X (gfx1250, wave32, WMMA).
//   Stage 1: V_j = u @ M_j            (48 GEMMs 4096x512x512, bf16 WMMA)
//   Stage 2: y  = sum_j Toeplitz(phi_j) @ V_j   (banded GEMM, inner dim L)
// Features:
//   * 2x4 output tiling per wave: every loaded B fragment feeds 2 WMMAs
//     (1-1.5 x b128 loads per WMMA), ping-pong double buffering on top
//   * TDM (tensor_load_to_lds, 6-arg form) stages the per-job filter LUT,
//     hand-built D# descriptor per ISA 8.3/8.4, s_wait_tensorcnt + barrier
// ---------------------------------------------------------------------------

#define L_      4096
#define K_      24
#define D_      512
#define JOBS    48          // 24 filters x {plus, minus}
#define JPP     8           // jobs per pass (bounds workspace)
#define NPASS   (JOBS / JPP)
#define PHI_PAD 4160        // 4096 + 64 zero pad (masks s > l for free)

#define USE_TDM 1
#if defined(__has_builtin)
#if __has_builtin(__builtin_amdgcn_tensor_load_to_lds) && \
    __has_builtin(__builtin_amdgcn_s_wait_tensorcnt) && USE_TDM
#define HAVE_TDM 1
#endif
#endif

typedef __attribute__((ext_vector_type(16))) __bf16 v16bf;
typedef __attribute__((ext_vector_type(8)))  __bf16 v8bf;
typedef __attribute__((ext_vector_type(8)))  float  v8f;
typedef __attribute__((ext_vector_type(4)))  unsigned int u32x4;
typedef __attribute__((ext_vector_type(8)))  int          i32x8;
typedef __attribute__((ext_vector_type(4)))  int          i32x4;

static __device__ __forceinline__ v16bf cat16(v8bf lo, v8bf hi) {
    return __builtin_shufflevector(lo, hi, 0,1,2,3,4,5,6,7,8,9,10,11,12,13,14,15);
}

static __device__ __forceinline__ v8f wmma_bf16(v16bf A, v16bf B, v8f C) {
    return __builtin_amdgcn_wmma_f32_16x16x32_bf16(false, A, false, B,
                                                   (short)0, C, false, false);
}

#ifdef HAVE_TDM
// Generic pointer to an LDS object: low 32 bits are the LDS byte address
// (ISA 10.2: LDS aperture address truncates to the in-wave LDS offset).
static __device__ __forceinline__ unsigned lds_addr_of(const void* p) {
    return (unsigned)(unsigned long long)(uintptr_t)p;
}

// 1-row tensor load: nelem bf16 elements, global -> LDS.  D# per ISA 8.3/8.4.
static __device__ __forceinline__
void tdm_load_row_to_lds(unsigned lds_addr, const void* gptr, unsigned nelem) {
    unsigned long long ga = (unsigned long long)(uintptr_t)gptr;
    u32x4 g0;
    g0[0] = 1u;                                    // count=1, user mode
    g0[1] = lds_addr;                              // lds_addr [63:32]
    g0[2] = (unsigned)(ga & 0xFFFFFFFFu);          // global_addr lo
    g0[3] = (unsigned)((ga >> 32) & 0x01FFFFFFu)   // global_addr hi (57-bit)
            | (2u << 30);                          // type=2 ("image")
    i32x8 g1;
    g1[0] = (int)(1u << 16);                       // data_size=1 -> 2 bytes
    g1[1] = (int)((nelem & 0xFFFFu) << 16);        // tensor_dim0[15:0] @ [31:16]
    g1[2] = (int)((nelem >> 16) | (1u << 16));     // tensor_dim0 hi | tensor_dim1=1
    g1[3] = (int)((nelem & 0xFFFFu) << 16);        // tile_dim0 @ [127:112]
    g1[4] = 1;                                     // tile_dim1=1, tile_dim2=0
    g1[5] = (int)nelem;                            // tensor_dim0_stride lo
    g1[6] = 0;
    g1[7] = 0;
    i32x4 gz4 = {0, 0, 0, 0};
    i32x8 gz8 = {0, 0, 0, 0, 0, 0, 0, 0};
    __builtin_amdgcn_tensor_load_to_lds(g0, g1, gz4, gz4, gz8, 0);
}
#endif

// ---- Stage 0a: u f32 -> bf16 ----------------------------------------------
__global__ __launch_bounds__(256)
void stu_cvt_u(const float* __restrict__ u, __bf16* __restrict__ ub) {
    int idx = blockIdx.x * 256 + threadIdx.x;
    if (idx < L_ * D_) ub[idx] = (__bf16)u[idx];
}

// ---- Stage 0b: M[k][d][o] f32 -> Mt[job][o][d] bf16 (tiled transpose) -----
__global__ __launch_bounds__(256)
void stu_cvt_M(const float* __restrict__ Mp, const float* __restrict__ Mm,
               __bf16* __restrict__ Mt) {
    __shared__ float tile[32][33];
    const int j = blockIdx.z;
    const int k = j % K_;
    const float* src = (j < K_ ? Mp : Mm) + (size_t)k * D_ * D_;   // [d][o]
    const int o0 = blockIdx.x * 32, d0 = blockIdx.y * 32;
    const int tx = threadIdx.x, ty = threadIdx.y;                  // (32, 8)
#pragma unroll
    for (int i = 0; i < 4; ++i)
        tile[ty + 8 * i][tx] = src[(size_t)(d0 + ty + 8 * i) * D_ + (o0 + tx)];
    __syncthreads();
    __bf16* dst = Mt + (size_t)j * D_ * D_;                        // [o][d]
#pragma unroll
    for (int i = 0; i < 4; ++i)
        dst[(size_t)(o0 + ty + 8 * i) * D_ + (d0 + tx)] = (__bf16)tile[tx][ty + 8 * i];
}

// ---- Stage 0c: reversed, sign-folded, zero-padded filter LUTs -------------
__global__ __launch_bounds__(256)
void stu_cvt_phi(const float* __restrict__ fl, __bf16* __restrict__ phr) {
    int idx = blockIdx.x * 256 + threadIdx.x;
    if (idx >= JOBS * PHI_PAD) return;
    int j = idx / PHI_PAD, i = idx % PHI_PAD;
    float v = 0.0f;
    if (i < L_) {
        int t = (L_ - 1) - i;
        v = fl[t * K_ + (j % K_)];
        if (j >= K_ && (t & 1)) v = -v;
    }
    phr[idx] = (__bf16)v;
}

// ---- Stage 1: Vt[job][o][s] = sum_d Mt[job][o][d] * u[s][d] ---------------
// Wave tile: 32 o x 64 s  (2 A-frags x 4 B-frags -> 8 WMMAs per K-chunk)
struct Frag1 { v16bf A[2]; v16bf B[4]; };

static __device__ __forceinline__
void s1_load(Frag1& f, const __bf16* __restrict__ Arow0,
             const __bf16* __restrict__ ubase, int d0, int h) {
#pragma unroll
    for (int a = 0; a < 2; ++a) {
        const __bf16* Ap = Arow0 + (size_t)a * 16 * D_ + d0 + 8 * h;
        f.A[a] = cat16(*reinterpret_cast<const v8bf*>(Ap),
                       *reinterpret_cast<const v8bf*>(Ap + 16));
    }
#pragma unroll
    for (int tt = 0; tt < 4; ++tt) {
        const __bf16* Bp = ubase + (size_t)tt * 16 * D_ + d0 + 8 * h;
        f.B[tt] = cat16(reinterpret_cast<const v8bf*>(Bp)[0],
                        reinterpret_cast<const v8bf*>(Bp)[2]);
    }
}

__global__ __launch_bounds__(256)
void stu_stage1(const __bf16* __restrict__ Mt_pass,  // [JPP][D][D]
                const __bf16* __restrict__ ub,       // [L][D]
                __bf16* __restrict__ Vt) {           // [JPP][D][L]
    const int tid  = threadIdx.x;
    const int wave = tid >> 5, lane = tid & 31;
    const int h = lane >> 4, ln = lane & 15;
    const int o0 = blockIdx.y * 64 + (wave & 1) * 32;
    const int s0 = blockIdx.x * 256 + (wave >> 1) * 64;
    const int jz = blockIdx.z;

    const __bf16* Arow0 = Mt_pass + (size_t)jz * D_ * D_ + (size_t)(o0 + ln) * D_;
    const __bf16* ubase = ub + (size_t)(s0 + ln) * D_;
    v8f acc[2][4] = {};

    Frag1 f0, f1;
    s1_load(f0, Arow0, ubase, 0, h);
#pragma unroll
    for (int it = 0; it < 16; ++it) {                   // d0 = it*32
        Frag1& cur = (it & 1) ? f1 : f0;
        Frag1& nxt = (it & 1) ? f0 : f1;
        if (it + 1 < 16) s1_load(nxt, Arow0, ubase, (it + 1) * 32, h);
#pragma unroll
        for (int a = 0; a < 2; ++a)
#pragma unroll
            for (int tt = 0; tt < 4; ++tt)
                acc[a][tt] = wmma_bf16(cur.A[a], cur.B[tt], acc[a][tt]);
    }
#pragma unroll
    for (int a = 0; a < 2; ++a)
#pragma unroll
        for (int tt = 0; tt < 4; ++tt)
#pragma unroll
            for (int v = 0; v < 8; ++v)
                Vt[(size_t)jz * D_ * L_ + (size_t)(o0 + a * 16 + v + 8 * h) * L_ +
                   (s0 + tt * 16 + ln)] = (__bf16)acc[a][tt][v];
}

// ---- Stage 2: y[l][o] (+)= sum_j sum_{s<=l} phi_j[l-s] * V_j[s][o] --------
// Wave tile: 32 l x 64 o  (2 Toeplitz A-frags x 4 B-frags -> 8 WMMAs/chunk)
static __device__ __forceinline__
void s2_loadB(v16bf* B, const __bf16* __restrict__ Vbase, int s0, int h) {
#pragma unroll
    for (int tt = 0; tt < 4; ++tt) {
        const __bf16* Bp = Vbase + (size_t)tt * 16 * L_ + s0 + 8 * h;
        B[tt] = cat16(reinterpret_cast<const v8bf*>(Bp)[0],
                      reinterpret_cast<const v8bf*>(Bp)[2]);
    }
}

static __device__ __forceinline__ v16bf s2_buildA(const __bf16* phs, int ib) {
    v16bf A;
#pragma unroll
    for (int j = 0; j < 8; ++j) { A[j] = phs[ib + j]; A[j + 8] = phs[ib + 16 + j]; }
    return A;
}

__global__ __launch_bounds__(128)
void stu_stage2(const __bf16* __restrict__ Vt,        // [JPP][D][L]
                const __bf16* __restrict__ phr_pass,  // [JPP][PHI_PAD]
                float* __restrict__ y, int first) {
    __shared__ __bf16 phs[PHI_PAD];
    const int tid  = threadIdx.x;
    const int wave = tid >> 5, lane = tid & 31;
    const int h = lane >> 4, ln = lane & 15;
    const int lt = (int)(gridDim.x - 1u - blockIdx.x);   // heavy tiles first
    const int l0 = lt * 32;                              // 32 rows per block
    const int o0 = blockIdx.y * 256 + wave * 64;
    const int ibase0 = (L_ - 1) - (l0 + ln);             // reversed-LUT base (a=0)
    const int smax = l0 + 31;

    v8f acc[2][4] = {};

    for (int jl = 0; jl < JPP; ++jl) {
        const __bf16* ph = phr_pass + (size_t)jl * PHI_PAD;
#ifdef HAVE_TDM
        if (wave == 0) {                      // wave-uniform branch; TDM ignores EXEC
            tdm_load_row_to_lds(lds_addr_of(&phs[0]), ph, PHI_PAD);
            __builtin_amdgcn_s_wait_tensorcnt(0);
        }
        __syncthreads();
#else
        for (int i = tid; i < PHI_PAD; i += 128) phs[i] = ph[i];
        __syncthreads();
#endif
        const __bf16* Vbase = Vt + (size_t)jl * D_ * L_ + (size_t)(o0 + ln) * L_;

        v16bf B0[4], B1[4];
        s2_loadB(B0, Vbase, 0, h);
        for (int s0 = 0; s0 <= smax; s0 += 64) {
            const bool has1 = (s0 + 32 <= smax);
            if (has1) s2_loadB(B1, Vbase, s0 + 32, h);
            __builtin_prefetch((const void*)(Vbase + s0 + 128 + 16 * h), 0, 0);
#pragma unroll
            for (int a = 0; a < 2; ++a) {
                v16bf A = s2_buildA(phs, ibase0 - 16 * a + s0 + 8 * h);
#pragma unroll
                for (int tt = 0; tt < 4; ++tt)
                    acc[a][tt] = wmma_bf16(A, B0[tt], acc[a][tt]);
            }
            if (has1) {
                if (s0 + 64 <= smax) s2_loadB(B0, Vbase, s0 + 64, h);
#pragma unroll
                for (int a = 0; a < 2; ++a) {
                    v16bf A2 = s2_buildA(phs, ibase0 - 16 * a + s0 + 32 + 8 * h);
#pragma unroll
                    for (int tt = 0; tt < 4; ++tt)
                        acc[a][tt] = wmma_bf16(A2, B1[tt], acc[a][tt]);
                }
            }
        }
        __syncthreads();   // protect phs before next job's staging
    }

#pragma unroll
    for (int a = 0; a < 2; ++a)
#pragma unroll
        for (int tt = 0; tt < 4; ++tt)
#pragma unroll
            for (int v = 0; v < 8; ++v) {
                size_t idx = (size_t)(l0 + a * 16 + v + 8 * h) * D_ +
                             (o0 + tt * 16 + ln);
                if (first) y[idx] = acc[a][tt][v];
                else       y[idx] += acc[a][tt][v];
            }
}

// ---------------------------------------------------------------------------
extern "C" void kernel_launch(void* const* d_in, const int* in_sizes, int n_in,
                              void* d_out, int out_size, void* d_ws, size_t ws_size,
                              hipStream_t stream) {
    (void)in_sizes; (void)n_in; (void)out_size;
    const float* u  = (const float*)d_in[0];
    const float* fl = (const float*)d_in[1];
    const float* Mp = (const float*)d_in[2];
    const float* Mm = (const float*)d_in[3];
    float* y = (float*)d_out;

    const size_t OFF_UB  = 0;                                   // L*D bf16     = 4 MB
    const size_t OFF_MT  = OFF_UB + (size_t)L_ * D_ * 2;        // 48*D*D bf16  = 25.2 MB
    const size_t OFF_PHR = OFF_MT + (size_t)JOBS * D_ * D_ * 2; // 48*4160 bf16 = 0.4 MB
    const size_t OFF_VT  = OFF_PHR + (size_t)JOBS * PHI_PAD * 2;// JPP*D*L bf16 = 33.6 MB
    const size_t NEED    = OFF_VT + (size_t)JPP * D_ * L_ * 2;  // ~63.3 MB
    if (ws_size < NEED) return;

    char* ws = (char*)d_ws;
    __bf16* ub  = (__bf16*)(ws + OFF_UB);
    __bf16* Mt  = (__bf16*)(ws + OFF_MT);
    __bf16* phr = (__bf16*)(ws + OFF_PHR);
    __bf16* Vt  = (__bf16*)(ws + OFF_VT);

    stu_cvt_u  <<<(L_ * D_ + 255) / 256, 256, 0, stream>>>(u, ub);
    stu_cvt_M  <<<dim3(16, 16, JOBS), dim3(32, 8), 0, stream>>>(Mp, Mm, Mt);
    stu_cvt_phi<<<(JOBS * PHI_PAD + 255) / 256, 256, 0, stream>>>(fl, phr);

    for (int p = 0; p < NPASS; ++p) {
        stu_stage1<<<dim3(16, 8, JPP), 256, 0, stream>>>(
            Mt + (size_t)p * JPP * D_ * D_, ub, Vt);
        stu_stage2<<<dim3(128, 2), 128, 0, stream>>>(
            Vt, phr + (size_t)p * JPP * PHI_PAD, y, p == 0);
    }
}
